// EncoderBlock_20194936226721
// MI455X (gfx1250) — matrix operand order repaired and checked by reading, not compiled, not run on hardware
//
#include <hip/hip_runtime.h>

// ---------------------------------------------------------------------------
// Types
// ---------------------------------------------------------------------------
typedef __bf16 bf16;
typedef __attribute__((ext_vector_type(16))) __bf16 v16bf;
typedef __attribute__((ext_vector_type(8)))  float  v8f;
typedef __attribute__((ext_vector_type(4)))  float  v4f;
typedef __attribute__((ext_vector_type(4)))  __bf16 v4bf;
typedef __attribute__((ext_vector_type(4)))  unsigned int u32x4;

union Frag {            // 8 VGPRs = 16 bf16 per lane
    v16bf v;
    u32x4 q[2];
};

// z -> element offset map:  off(z) = (z / div) * s1 + (z % mod) * s2
struct ZMap { int div; int mod; long long s1; long long s2; };
__device__ __forceinline__ long long zoff(ZMap m, int z) {
    return (long long)(z / m.div) * m.s1 + (long long)(z % m.mod) * m.s2;
}

// gfx1250 async global->LDS copy of 16 bytes (ASYNCcnt-tracked, no VGPR data).
__device__ __forceinline__ void async_copy_b128(const void* gptr, void* lptr) {
    unsigned lds = (unsigned)(unsigned long long)lptr;        // LDS offset = addr[31:0]
    unsigned long long g = (unsigned long long)gptr;
    asm volatile("global_load_async_to_lds_b128 %0, %1, off"
                 :: "v"(lds), "v"(g) : "memory");
}
__device__ __forceinline__ void wait_async0() {
    asm volatile("s_wait_asynccnt 0x0" ::: "memory");
}

// ---------------------------------------------------------------------------
// fp32 -> bf16 conversion (vectorized 4-wide)
// ---------------------------------------------------------------------------
__global__ __launch_bounds__(256) void cvt_f32_bf16(const float* __restrict__ in,
                                                    bf16* __restrict__ out,
                                                    long long n4) {
    long long i = (long long)blockIdx.x * 256 + threadIdx.x;
    long long stride = (long long)gridDim.x * 256;
    for (; i < n4; i += stride) {
        v4f f = *(const v4f*)(in + i * 4);
        v4bf o;
        o.x = (bf16)f.x; o.y = (bf16)f.y; o.z = (bf16)f.z; o.w = (bf16)f.w;
        *(v4bf*)(out + i * 4) = o;
    }
}

// ---------------------------------------------------------------------------
// Tiled bf16 WMMA GEMM, async double-buffered through LDS.
//   Block = 256 threads = 8 waves, wave tile = 32(M) x 64(N) = 2x4 WMMA tiles.
//   BM = WROWS*32, BN = WCOLS*64, K-step = 32, WROWS*WCOLS == 8.
//   BT=false: B is [K,N] row-major (transposed into LDS synchronously).
//   BT=true : B source is [N,K] row-major (A*B^T) -> async straight copy.
// Epilogue: v = acc*alpha + bias[n]; optional relu; write f32 and/or bf16.
// ---------------------------------------------------------------------------
template <int WROWS, int WCOLS, bool BT>
__global__ __launch_bounds__(256) void gemm_wmma(
    const bf16* __restrict__ A, const bf16* __restrict__ B,
    const float* __restrict__ bias, float* __restrict__ Cf, bf16* __restrict__ Cb,
    int K, int lda, int ldb, int ldc, float alpha, int relu,
    ZMap za, ZMap zb, ZMap zc) {

    constexpr int BM = WROWS * 32;
    constexpr int BN = WCOLS * 64;
    constexpr int LP = 40;                 // 32 + 8 bf16 pad per row

    __shared__ bf16 As[2][BM][LP];
    __shared__ bf16 Bs[2][BN][LP];         // N-major: Bs[buf][n][k]

    const int z = blockIdx.z;
    A += zoff(za, z);
    B += zoff(zb, z);
    const long long cbase = zoff(zc, z);

    const int tid    = threadIdx.x;
    const int lane   = tid & 31;
    const int w      = tid >> 5;
    const int wr     = w % WROWS;
    const int wc     = w / WROWS;
    const int laneLo = lane & 15;
    const int laneHi = lane >> 4;
    const int m0     = blockIdx.y * BM;
    const int n0     = blockIdx.x * BN;

    // ---- tile staging -----------------------------------------------------
    auto stage = [&](int kt, int buf) {
        const int k0 = kt << 5;
        // A tile (BM x 32): async 16B chunks straight into LDS
#pragma unroll
        for (int i = 0; i < BM / 64; ++i) {
            int c  = tid + i * 256;
            int r  = c >> 2;
            int ks = (c & 3) << 3;
            async_copy_b128(A + (long long)(m0 + r) * lda + k0 + ks,
                            &As[buf][r][ks]);
        }
        if (BT) {
            // B source already [N,K] row-major: async straight copy
#pragma unroll
            for (int i = 0; i < BN / 64; ++i) {
                int c  = tid + i * 256;
                int r  = c >> 2;
                int ks = (c & 3) << 3;
                async_copy_b128(B + (long long)(n0 + r) * ldb + k0 + ks,
                                &Bs[buf][r][ks]);
            }
        } else {
            // B is [K,N] row-major: transpose bf16 pairs into Bs[n][k]
#pragma unroll
            for (int i = 0; i < BN / 16; ++i) {
                int c  = tid + i * 256;
                int k  = c / (BN / 2);
                int np = c % (BN / 2);
                unsigned int val =
                    *(const unsigned int*)(B + (long long)(k0 + k) * ldb + n0 + np * 2);
                *(unsigned short*)&Bs[buf][2 * np + 0][k] = (unsigned short)(val & 0xffffu);
                *(unsigned short*)&Bs[buf][2 * np + 1][k] = (unsigned short)(val >> 16);
            }
        }
    };

    v8f acc[2][4];
    const v8f vzero = {0.f, 0.f, 0.f, 0.f, 0.f, 0.f, 0.f, 0.f};
#pragma unroll
    for (int i = 0; i < 2; ++i)
#pragma unroll
        for (int j = 0; j < 4; ++j) acc[i][j] = vzero;

    const int nK = K >> 5;
    stage(0, 0);                                   // prologue

    for (int kt = 0; kt < nK; ++kt) {
        const int cur = kt & 1;

        wait_async0();          // my async portion of tile kt has landed
        __syncthreads();        // everyone's portion landed; prev reads done

        if (kt + 1 < nK) stage(kt + 1, cur ^ 1);   // overlap fetch with math

        // ---- fragments per ISA 16-bit A(16x32) / B(32x16) layouts ----
        Frag a[2], b[4];
#pragma unroll
        for (int mt = 0; mt < 2; ++mt) {
            int row = wr * 32 + mt * 16 + laneLo;
            a[mt].q[0] = *(const u32x4*)&As[cur][row][laneHi * 8];      // K 0..7 / 8..15
            a[mt].q[1] = *(const u32x4*)&As[cur][row][16 + laneHi * 8]; // K 16..23 / 24..31
        }
#pragma unroll
        for (int nt = 0; nt < 4; ++nt) {
            int n = wc * 64 + nt * 16 + laneLo;
            b[nt].q[0] = *(const u32x4*)&Bs[cur][n][laneHi * 8];        // K 0..7 / 8..15
            b[nt].q[1] = *(const u32x4*)&Bs[cur][n][16 + laneHi * 8];   // K 16..23 / 24..31
        }

#pragma unroll
        for (int mt = 0; mt < 2; ++mt)
#pragma unroll
            for (int nt = 0; nt < 4; ++nt)
                acc[mt][nt] = __builtin_amdgcn_wmma_f32_16x16x32_bf16(
                    false, a[mt].v, false, b[nt].v, (short)0, acc[mt][nt],
                    false, false);
    }

    // ---- epilogue: C/D layout = 8 VGPRs, N=laneLo, M = r + laneHi*8 ----
#pragma unroll
    for (int mt = 0; mt < 2; ++mt) {
#pragma unroll
        for (int nt = 0; nt < 4; ++nt) {
            int gn = n0 + wc * 64 + nt * 16 + laneLo;
            float bv = bias ? bias[gn] : 0.f;
#pragma unroll
            for (int r = 0; r < 8; ++r) {
                int gm = m0 + wr * 32 + mt * 16 + laneHi * 8 + r;
                float v = acc[mt][nt][r] * alpha + bv;
                if (relu) v = fmaxf(v, 0.f);
                long long idx = cbase + (long long)gm * ldc + gn;
                if (Cf) Cf[idx] = v;
                if (Cb) Cb[idx] = (bf16)v;
            }
        }
    }
}

// ---------------------------------------------------------------------------
// In-place row softmax on bf16 [rows x rowLen], rowLen/256 elems per thread.
// ---------------------------------------------------------------------------
__global__ __launch_bounds__(256) void softmax_rows(bf16* __restrict__ a, int rowLen) {
    __shared__ float red[256];
    const int tid = threadIdx.x;
    const long long base = (long long)blockIdx.x * rowLen;
    const int per = rowLen >> 8;   // 8 for rowLen=2048

    float lv[8];
    float m = -3.4e38f;
    for (int i = 0; i < per; ++i) {
        lv[i] = (float)a[base + tid + i * 256];
        m = fmaxf(m, lv[i]);
    }
    red[tid] = m; __syncthreads();
    for (int s = 128; s > 0; s >>= 1) {
        if (tid < s) red[tid] = fmaxf(red[tid], red[tid + s]);
        __syncthreads();
    }
    m = red[0]; __syncthreads();

    float sum = 0.f;
    for (int i = 0; i < per; ++i) {
        lv[i] = __expf(lv[i] - m);
        sum += lv[i];
    }
    red[tid] = sum; __syncthreads();
    for (int s = 128; s > 0; s >>= 1) {
        if (tid < s) red[tid] += red[tid + s];
        __syncthreads();
    }
    const float inv = 1.0f / red[0];

    for (int i = 0; i < per; ++i)
        a[base + tid + i * 256] = (bf16)(lv[i] * inv);
}

// ---------------------------------------------------------------------------
// y = LayerNorm(a + resid) * g + b  ; writes f32 (always) and bf16 (optional).
// One block per row of D=1024, 4 elems/thread.
// ---------------------------------------------------------------------------
__global__ __launch_bounds__(256) void add_layernorm(
    const float* __restrict__ a, const float* __restrict__ resid,
    const float* __restrict__ g, const float* __restrict__ beta,
    float* __restrict__ outF, bf16* __restrict__ outB, int D) {
    __shared__ float red[256];
    const int tid = threadIdx.x;
    const long long base = (long long)blockIdx.x * D;
    const int per = D >> 8;   // 4

    float lv[4];
    float s = 0.f, s2 = 0.f;
    for (int i = 0; i < per; ++i) {
        int col = tid + i * 256;
        lv[i] = a[base + col] + resid[base + col];
        s += lv[i];
        s2 += lv[i] * lv[i];
    }
    red[tid] = s; __syncthreads();
    for (int t = 128; t > 0; t >>= 1) {
        if (tid < t) red[tid] += red[tid + t];
        __syncthreads();
    }
    s = red[0]; __syncthreads();
    red[tid] = s2; __syncthreads();
    for (int t = 128; t > 0; t >>= 1) {
        if (tid < t) red[tid] += red[tid + t];
        __syncthreads();
    }
    s2 = red[0];

    const float mean = s / (float)D;
    const float var  = s2 / (float)D - mean * mean;
    const float rs   = rsqrtf(var + 1e-5f);

    for (int i = 0; i < per; ++i) {
        int col = tid + i * 256;
        float v = (lv[i] - mean) * rs * g[col] + beta[col];
        outF[base + col] = v;
        if (outB) outB[base + col] = (bf16)v;
    }
}

// ---------------------------------------------------------------------------
// Launcher
// ---------------------------------------------------------------------------
extern "C" void kernel_launch(void* const* d_in, const int* in_sizes, int n_in,
                              void* d_out, int out_size, void* d_ws, size_t ws_size,
                              hipStream_t stream) {
    (void)in_sizes; (void)n_in; (void)out_size; (void)ws_size;

    constexpr int BATCH = 2, S = 2048, D = 1024, H = 16, DK = 64, DVH = 64,
                  DV = 1024, DF = 4096;
    constexpr long long ROWS = (long long)BATCH * S;      // 4096

    const float* x     = (const float*)d_in[0];
    const float* Wq    = (const float*)d_in[1];
    const float* Wk    = (const float*)d_in[2];
    const float* Wv    = (const float*)d_in[3];
    const float* Wo    = (const float*)d_in[4];
    const float* ln1_g = (const float*)d_in[5];
    const float* ln1_b = (const float*)d_in[6];
    const float* W1    = (const float*)d_in[7];
    const float* b1    = (const float*)d_in[8];
    const float* W2    = (const float*)d_in[9];
    const float* b2    = (const float*)d_in[10];
    const float* ln2_g = (const float*)d_in[11];
    const float* ln2_b = (const float*)d_in[12];
    float* out = (float*)d_out;

    // ---- workspace bump allocator (256B aligned) ----
    char* p = (char*)d_ws;
    auto alloc = [&](long long bytes) -> void* {
        void* r = (void*)p;
        p += (bytes + 255) & ~255ll;
        return r;
    };
    bf16*  xb   = (bf16*) alloc(ROWS * D * 2);
    bf16*  Wqb  = (bf16*) alloc((long long)H * D * DK * 2);
    bf16*  Wkb  = (bf16*) alloc((long long)H * D * DK * 2);
    bf16*  Wvb  = (bf16*) alloc((long long)H * D * DVH * 2);
    bf16*  Wob  = (bf16*) alloc((long long)DV * D * 2);
    bf16*  W1b  = (bf16*) alloc((long long)D * DF * 2);
    bf16*  W2b  = (bf16*) alloc((long long)DF * D * 2);
    bf16*  qb   = (bf16*) alloc((long long)BATCH * H * S * DK * 2);
    bf16*  kb   = (bf16*) alloc((long long)BATCH * H * S * DK * 2);
    bf16*  vb   = (bf16*) alloc((long long)BATCH * H * S * DVH * 2);
    bf16*  attn = (bf16*) alloc((long long)BATCH * H * S * S * 2);   // 256 MB
    bf16*  ob   = (bf16*) alloc(ROWS * DV * 2);
    float* aout = (float*)alloc(ROWS * D * 4);
    float* x1f  = (float*)alloc(ROWS * D * 4);
    bf16*  x1b  = (bf16*) alloc(ROWS * D * 2);
    bf16*  h1b  = (bf16*) alloc(ROWS * (long long)DF * 2);
    float* ffnf = (float*)alloc(ROWS * D * 4);

    auto cvt = [&](const float* in, bf16* o, long long n) {
        long long n4 = n >> 2;
        unsigned g = (unsigned)((n4 + 255) / 256);
        cvt_f32_bf16<<<g, 256, 0, stream>>>(in, o, n4);
    };

    // ---- 1) fp32 -> bf16 staging ----
    cvt(x,  xb,  ROWS * D);
    cvt(Wq, Wqb, (long long)H * D * DK);
    cvt(Wk, Wkb, (long long)H * D * DK);
    cvt(Wv, Wvb, (long long)H * D * DVH);
    cvt(Wo, Wob, (long long)DV * D);
    cvt(W1, W1b, (long long)D * DF);
    cvt(W2, W2b, (long long)DF * D);

    const ZMap Z0   = {1, 1, 0, 0};
    const ZMap zaP  = {H, 1, (long long)S * D, 0};               // A = x[b]
    const ZMap zbP  = {1 << 30, H, 0, (long long)D * DK};        // B = W[h]
    const ZMap zcP  = {1, 1, (long long)S * DK, 0};              // C = q/k/v[z]
    const ZMap zSK  = {1, 1, (long long)S * DK, 0};
    const ZMap zSS  = {1, 1, (long long)S * S, 0};
    const ZMap zcO  = {H, H, (long long)S * DV, (long long)DVH}; // o_concat slot

    // ---- 2) per-head Q/K/V projections: [2048x1024] x [1024x64], z = b*H+h
    gemm_wmma<8, 1, false><<<dim3(1, 8, 32), 256, 0, stream>>>(
        xb, Wqb, nullptr, nullptr, qb, D, D, DK, DK, 1.f, 0, zaP, zbP, zcP);
    gemm_wmma<8, 1, false><<<dim3(1, 8, 32), 256, 0, stream>>>(
        xb, Wkb, nullptr, nullptr, kb, D, D, DK, DK, 1.f, 0, zaP, zbP, zcP);
    gemm_wmma<8, 1, false><<<dim3(1, 8, 32), 256, 0, stream>>>(
        xb, Wvb, nullptr, nullptr, vb, D, D, DVH, DVH, 1.f, 0, zaP, zbP, zcP);

    // ---- 3) scores = Q * K^T / 8 : [2048x64] x [2048x64]^T, bf16 out ----
    gemm_wmma<4, 2, true><<<dim3(16, 16, 32), 256, 0, stream>>>(
        qb, kb, nullptr, nullptr, attn, DK, DK, DK, S, 0.125f, 0, zSK, zSK, zSS);

    // ---- 4) softmax in place over rows of 2048 ----
    softmax_rows<<<BATCH * H * S, 256, 0, stream>>>(attn, S);

    // ---- 5) o = attn * V : [2048x2048] x [2048x64] -> concat heads ----
    gemm_wmma<8, 1, false><<<dim3(1, 8, 32), 256, 0, stream>>>(
        attn, vb, nullptr, nullptr, ob, S, S, DVH, DV, 1.f, 0, zSS, zSK, zcO);

    // ---- 6) attn_out = o_concat @ Wo : [4096x1024] x [1024x1024] ----
    gemm_wmma<4, 2, false><<<dim3(8, 32, 1), 256, 0, stream>>>(
        ob, Wob, nullptr, aout, nullptr, DV, DV, D, D, 1.f, 0, Z0, Z0, Z0);

    // ---- 7) x1 = LN(attn_out + x) ----
    add_layernorm<<<(unsigned)ROWS, 256, 0, stream>>>(
        aout, x, ln1_g, ln1_b, x1f, x1b, D);

    // ---- 8) h1 = relu(x1 @ W1 + b1) : [4096x1024] x [1024x4096] ----
    gemm_wmma<4, 2, false><<<dim3(32, 32, 1), 256, 0, stream>>>(
        x1b, W1b, b1, nullptr, h1b, D, D, DF, DF, 1.f, 1, Z0, Z0, Z0);

    // ---- 9) ffn = h1 @ W2 + b2 : [4096x4096] x [4096x1024] ----
    gemm_wmma<4, 2, false><<<dim3(8, 32, 1), 256, 0, stream>>>(
        h1b, W2b, b2, ffnf, nullptr, DF, DF, D, D, 1.f, 0, Z0, Z0, Z0);

    // ---- 10) out = LN(ffn + x1) ----
    add_layernorm<<<(unsigned)ROWS, 256, 0, stream>>>(
        ffnf, x1f, ln2_g, ln2_b, out, nullptr, D);
}
